// MambaSSM_53953379172830
// MI455X (gfx1250) — matrix operand order repaired and checked by reading, not compiled, not run on hardware
//
#include <hip/hip_runtime.h>
#include <hip/hip_bf16.h>
#include <math.h>
#include <stdint.h>

// ---- problem constants (match reference) ----
#define D_MODEL 512
#define D_INNER 1024
#define D_STATE 128
#define D_CONV  4
#define DT_RANK 32
#define BATCH   2
#define SEQ     1024
#define DBL_W   (DT_RANK + 2 * D_STATE)   // 288
#define CHUNK   16                        // timesteps per TDM slab

typedef __attribute__((ext_vector_type(16))) __bf16 v16bf;
typedef __attribute__((ext_vector_type(8)))  float  v8f;
typedef __attribute__((ext_vector_type(4)))  unsigned int v4u;
typedef __attribute__((ext_vector_type(8)))  int    v8i;
typedef __attribute__((ext_vector_type(4)))  int    v4i;

// -------------------------------------------------------------------------
// Pack fp32 weights [K,N] row-major -> bf16 WMMA B-fragment order:
//   tile (kTile,nTile) of shape 32x16; per lane l (half=l>>4, l16=l&15)
//   element i (0..15) is W[32*kTile + 16*half + i, 16*nTile + l16].
// Each lane's 16 bf16 land contiguously => GEMM loads one 32B vector.
// -------------------------------------------------------------------------
__global__ void mamba_pack_bf16(const float* __restrict__ W, int N,
                                __bf16* __restrict__ P, int total) {
  int i = blockIdx.x * blockDim.x + threadIdx.x;   // total = K*N
  if (i >= total) return;
  const int e    = i & 15;
  const int lane = (i >> 4) & 31;
  const int tile = i >> 9;                 // kTile * tilesN + nTile
  const int tilesN = N >> 4;
  const int kTile = tile / tilesN;
  const int nTile = tile % tilesN;
  const int k = kTile * 32 + ((lane >> 4) << 4) + e;
  const int n = nTile * 16 + (lane & 15);
  P[i] = (__bf16)W[(long)k * N + n];
}

// -------------------------------------------------------------------------
// C[M,N] = act( A[M,K](f32, lda) * Bp(packed bf16 fragments) + bias[N] )
// One wave per 16 x (16*SUBT) strip; SUBT accumulators share one A fragment;
// v_wmma_f32_16x16x32_bf16 over K in steps of 32. SUBT is compile-time so the
// WMMA sequence is branch-free (no exec juggling between matrix ops).
// act: 0 = none, 1 = clip(softplus(v + 1e-4), 1e-3, 1e-1)  (dt path)
// -------------------------------------------------------------------------
template <int SUBT>
__global__ __launch_bounds__(256)
void mamba_wmma_gemm(const float* __restrict__ A, int lda,
                     const __bf16* __restrict__ Bpk,
                     float* __restrict__ C, int ldc,
                     int M, int N, int K,
                     int stripCol0,      // first column tile (16-col units)
                     int stripsPerRow,   // SUBT-strips per row in this launch
                     const float* __restrict__ bias, int act) {
  const int lane = threadIdx.x & 31;
  const int wid  = threadIdx.x >> 5;
  const int tilesN = N >> 4;
  const int work = (M >> 4) * stripsPerRow;
  const int t = blockIdx.x * 8 + wid;
  if (t >= work) return;
  const int tm  = (t / stripsPerRow) << 4;
  const int nt0 = stripCol0 + (t % stripsPerRow) * SUBT;  // first 16-col tile

  const int half = lane >> 4;   // this code's order: lanes 0-15 take K 0..15, lanes 16-31 K 16..31
  const int l16  = lane & 15;

  v8f acc[SUBT];
  #pragma unroll
  for (int jj = 0; jj < SUBT; ++jj) acc[jj] = (v8f){};

  const float* Arow = A + (long)(tm + l16) * lda;    // this lane's A row
  const v16bf* Bp   = (const v16bf*)Bpk;             // one fragment per lane

  const int KT = K >> 5;
  for (int kt = 0; kt < KT; ++kt) {
    const float* Ak = Arow + (kt << 5);
    if (kt + 1 < KT) __builtin_prefetch(Ak + 32, 0, 1);   // global_prefetch_b8

    // A fragment: 16x32 bf16; lanes 0-15: K 0..15, lanes 16-31: K 16..31
    const float4 a0 = *(const float4*)(Ak + half * 16);
    const float4 a1 = *(const float4*)(Ak + half * 16 + 4);
    const float4 a2 = *(const float4*)(Ak + half * 16 + 8);
    const float4 a3 = *(const float4*)(Ak + half * 16 + 12);
    v16bf a;
    a[0]  = (__bf16)a0.x; a[1]  = (__bf16)a0.y; a[2]  = (__bf16)a0.z; a[3]  = (__bf16)a0.w;
    a[4]  = (__bf16)a1.x; a[5]  = (__bf16)a1.y; a[6]  = (__bf16)a1.z; a[7]  = (__bf16)a1.w;
    a[8]  = (__bf16)a2.x; a[9]  = (__bf16)a2.y; a[10] = (__bf16)a2.z; a[11] = (__bf16)a2.w;
    a[12] = (__bf16)a3.x; a[13] = (__bf16)a3.y; a[14] = (__bf16)a3.z; a[15] = (__bf16)a3.w;

    const long bbase = (long)((kt * tilesN + nt0) << 5) + lane;
    #pragma unroll
    for (int jj = 0; jj < SUBT; ++jj) {
      const v16bf b = Bp[bbase + (jj << 5)];   // one aligned 32B fragment load
      acc[jj] = __builtin_amdgcn_wmma_f32_16x16x32_bf16(
          false, a, false, b, (short)0, acc[jj], false, false);
    }
  }

  #pragma unroll
  for (int jj = 0; jj < SUBT; ++jj) {
    const int tn = (nt0 + jj) << 4;
    const float bn = bias ? bias[tn + l16] : 0.0f;
    #pragma unroll
    for (int r = 0; r < 8; ++r) {
      float v = acc[jj][r] + bn;             // C layout: VGPR r -> row 8*half + r
      if (act == 1) {
        float xf = v + 1e-4f;                // dt floor
        float sp = (xf > 20.0f) ? xf : log1pf(__expf(xf));
        v = fminf(0.1f, fmaxf(0.001f, sp));
      }
      C[(long)(tm + half * 8 + r) * ldc + tn + l16] = v;
    }
  }
}

// -------------------------------------------------------------------------
// depthwise causal conv width 4 + bias: u[b,s,d] from xs = xr[..., 0:D_INNER]
// -------------------------------------------------------------------------
__global__ void mamba_conv1d(const float* __restrict__ xr,
                             const float* __restrict__ w,
                             const float* __restrict__ bvec,
                             float* __restrict__ u) {
  int i = blockIdx.x * blockDim.x + threadIdx.x;   // over BATCH*SEQ*D_INNER
  if (i >= BATCH * SEQ * D_INNER) return;
  int d  = i & (D_INNER - 1);
  int s  = (i >> 10) & (SEQ - 1);
  int bb = i >> 20;
  const float* xs = xr + (long)bb * SEQ * (2 * D_INNER) + d;
  float acc = bvec[d];
  #pragma unroll
  for (int j = 0; j < D_CONV; ++j) {
    int sp = s - (D_CONV - 1) + j;
    if (sp >= 0) acc += w[d * D_CONV + j] * xs[(long)sp * (2 * D_INNER)];
  }
  u[i] = acc;
}

// -------------------------------------------------------------------------
// TDM: DMA a [CHUNK x 256] f32 tile (B_t||C_t rows, stride DBL_W) into LDS.
// D# per ISA 8.3/8.4: group0 = {count, lds_addr, global_addr, type=2};
// group1 = {data_size=4B, tensor_dim0/1, tile_dim0/1, dim0_stride}.
// -------------------------------------------------------------------------
__device__ __forceinline__ void tdm_load_bc(const float* gsrc, void* ldsdst) {
  unsigned lds = (unsigned)(unsigned long long)(uintptr_t)ldsdst;
  unsigned long long ga = (unsigned long long)(uintptr_t)gsrc;
  v4u g0 = { 1u,                                   // count=1, user descriptor
             lds,                                  // [63:32]  lds_addr (bytes)
             (unsigned)ga,                         // [95:64]  global_addr lo
             (unsigned)((ga >> 32) & 0x1FFFFFFu) | (2u << 30) };  // addr hi | type=2
  v8i g1 = { (int)0x00020000u,                     // data_size=2 (4 bytes)
             (int)(256u << 16),                    // tensor_dim0 = 256 elems
             (int)((unsigned)CHUNK << 16),         // tensor_dim1 = CHUNK rows
             (int)(256u << 16),                    // tile_dim0 = 256
             (int)CHUNK,                           // tile_dim1 = CHUNK
             (int)DBL_W,                           // tensor_dim0_stride = 288
             0, 0 };
  v4i gz = { 0, 0, 0, 0 };                         // 2D tensor: groups 2/3 unused
#if defined(__clang_major__) && (__clang_major__ >= 23)
  v8i gz8 = { 0, 0, 0, 0, 0, 0, 0, 0 };
  __builtin_amdgcn_tensor_load_to_lds(g0, g1, gz, gz, gz8, 0);
#else
  __builtin_amdgcn_tensor_load_to_lds(g0, g1, gz, gz, 0);
#endif
}

// -------------------------------------------------------------------------
// selective scan. block = 256 thr = 8 waves; each wave owns one d-channel,
// 4 states per lane. B_t/C_t slabs DMA'd by the TDM into double-buffered
// LDS (wave 0 issues, s_wait_tensorcnt gates), overlapped with compute.
// Residual (xr second half) fused into the output.
// -------------------------------------------------------------------------
__global__ __launch_bounds__(256)
void mamba_scan(const float* __restrict__ u,
                const float* __restrict__ dt,
                const float* __restrict__ dbl,
                const float* __restrict__ xr,
                float* __restrict__ yres) {
  __shared__ float sBC[2][CHUNK * 256];            // 32 KB of the 320 KB WGP LDS
  const int lane = threadIdx.x & 31;
  const int wid  = threadIdx.x >> 5;
  const int bb   = blockIdx.x >> 7;                // / (D_INNER/8)
  const int d    = ((blockIdx.x & 127) << 3) + wid;
  const int n0   = lane << 2;

  float h0 = 0.f, h1 = 0.f, h2 = 0.f, h3 = 0.f;
  const long rowBC = (long)bb * SEQ;
  const float* bcBase = dbl + rowBC * DBL_W + DT_RANK;   // B||C slab of row 0

  if (wid == 0) tdm_load_bc(bcBase, &sBC[0][0]);         // prologue: chunk 0

  for (int c = 0; c < SEQ / CHUNK; ++c) {
    if (wid == 0) __builtin_amdgcn_s_wait_tensorcnt(0);  // chunk c landed
    __syncthreads();
    if (wid == 0 && c + 1 < SEQ / CHUNK)                 // prefetch chunk c+1
      tdm_load_bc(bcBase + (long)(c + 1) * CHUNK * DBL_W, &sBC[(c + 1) & 1][0]);

    const float* bc = &sBC[c & 1][0];
    for (int r = 0; r < CHUNK; ++r) {
      const int t = c * CHUNK + r;
      const long idx = (rowBC + t) * D_INNER + d;
      const float xt  = u[idx];
      const float dtt = dt[idx];
      const float om  = 1.0f - dtt;
      const float* Bv = bc + r * 256;        // [0:128) = B_t, [128:256) = C_t
      h0 = h0 * om + xt * Bv[n0 + 0];
      h1 = h1 * om + xt * Bv[n0 + 1];
      h2 = h2 * om + xt * Bv[n0 + 2];
      h3 = h3 * om + xt * Bv[n0 + 3];
      float p = h0 * Bv[128 + n0 + 0] + h1 * Bv[128 + n0 + 1]
              + h2 * Bv[128 + n0 + 2] + h3 * Bv[128 + n0 + 3];
      #pragma unroll
      for (int off = 16; off > 0; off >>= 1) p += __shfl_xor(p, off, 32);
      if (lane == 0)
        yres[idx] = p + xr[(rowBC + t) * (2 * D_INNER) + D_INNER + d];
    }
    __syncthreads();                         // chunk consumed; buffer reusable
  }
}

// -------------------------------------------------------------------------
extern "C" void kernel_launch(void* const* d_in, const int* in_sizes, int n_in,
                              void* d_out, int out_size, void* d_ws, size_t ws_size,
                              hipStream_t stream) {
  (void)in_sizes; (void)n_in; (void)out_size; (void)ws_size;
  const float* x      = (const float*)d_in[0];
  const float* W_in   = (const float*)d_in[1];
  const float* conv_w = (const float*)d_in[2];
  const float* conv_b = (const float*)d_in[3];
  const float* W_x    = (const float*)d_in[4];
  const float* W_dt   = (const float*)d_in[5];
  const float* b_dt   = (const float*)d_in[6];
  const float* W_out  = (const float*)d_in[7];
  float* out = (float*)d_out;

  char* p = (char*)d_ws;
  auto carve = [&](size_t bytes) {
    char* r = p; p += (bytes + 255) & ~(size_t)255; return r;
  };
  __bf16* wInB  = (__bf16*)carve(sizeof(__bf16) * D_MODEL * 2 * D_INNER);
  __bf16* wXB   = (__bf16*)carve(sizeof(__bf16) * D_INNER * DBL_W);
  __bf16* wDtB  = (__bf16*)carve(sizeof(__bf16) * DT_RANK * D_INNER);
  __bf16* wOutB = (__bf16*)carve(sizeof(__bf16) * D_INNER * D_MODEL);
  float* xr   = (float*)carve(sizeof(float) * BATCH * SEQ * 2 * D_INNER);
  float* u    = (float*)carve(sizeof(float) * BATCH * SEQ * D_INNER);
  float* dbl  = (float*)carve(sizeof(float) * BATCH * SEQ * DBL_W);
  float* dtb  = (float*)carve(sizeof(float) * BATCH * SEQ * D_INNER);
  float* yres = (float*)carve(sizeof(float) * BATCH * SEQ * D_INNER);

  const int M = BATCH * SEQ;   // 2048 rows for all projections

  // 1. weight packing into WMMA fragment order (bf16)
  {
    int n;
    n = D_MODEL * 2 * D_INNER;
    mamba_pack_bf16<<<(n + 255) / 256, 256, 0, stream>>>(W_in, 2 * D_INNER, wInB, n);
    n = D_INNER * DBL_W;
    mamba_pack_bf16<<<(n + 255) / 256, 256, 0, stream>>>(W_x, DBL_W, wXB, n);
    n = DT_RANK * D_INNER;
    mamba_pack_bf16<<<(n + 255) / 256, 256, 0, stream>>>(W_dt, D_INNER, wDtB, n);
    n = D_INNER * D_MODEL;
    mamba_pack_bf16<<<(n + 255) / 256, 256, 0, stream>>>(W_out, D_MODEL, wOutB, n);
  }

  auto gemm = [&](const float* A, int lda, const __bf16* Bpk,
                  float* C, int ldc, int N, int K,
                  const float* bias, int act) {
    const int full = N >> 6;                         // 64-wide strips
    if (full > 0) {
      int work = (M >> 4) * full;
      mamba_wmma_gemm<4><<<(work + 7) / 8, 256, 0, stream>>>(
          A, lda, Bpk, C, ldc, M, N, K, 0, full, bias, act);
    }
    if (N & 63) {                                    // 32-wide edge (N=288 case)
      int work = M >> 4;
      mamba_wmma_gemm<2><<<(work + 7) / 8, 256, 0, stream>>>(
          A, lda, Bpk, C, ldc, M, N, K, full << 2, 1, bias, act);
    }
  };

  // 2. xr = x @ W_in                        [2048,512] x [512,2048]
  gemm(x, D_MODEL, wInB, xr, 2 * D_INNER, 2 * D_INNER, D_MODEL, nullptr, 0);

  // 3. depthwise causal conv + bias -> u
  {
    int n = BATCH * SEQ * D_INNER;
    mamba_conv1d<<<(n + 255) / 256, 256, 0, stream>>>(xr, conv_w, conv_b, u);
  }

  // 4. dbl = u @ W_x                        [2048,1024] x [1024,288]
  gemm(u, D_INNER, wXB, dbl, DBL_W, DBL_W, D_INNER, nullptr, 0);

  // 5. dt = clip(softplus(dbl[:, :32] @ W_dt + b_dt + floor))   K=32, fused act
  gemm(dbl, DBL_W, wDtB, dtb, D_INNER, D_INNER, DT_RANK, b_dt, 1);

  // 6. selective scan (TDM double-buffered B/C) + residual -> yres
  mamba_scan<<<BATCH * (D_INNER / 8), 256, 0, stream>>>(u, dtb, dbl, xr, yres);

  // 7. out = yres @ W_out                   [2048,1024] x [1024,512]
  gemm(yres, D_INNER, wOutB, out, D_MODEL, D_MODEL, D_INNER, nullptr, 0);
}
